// ExternalAttentionLayer_88072599372329
// MI455X (gfx1250) — compile-verified, matrix operand order checked by reading, not run on hardware
//
#include <hip/hip_runtime.h>

typedef __attribute__((ext_vector_type(16))) _Float16 v16h;
typedef __attribute__((ext_vector_type(8)))  _Float16 v8h;
typedef __attribute__((ext_vector_type(8)))  float    v8f;
typedef __attribute__((ext_vector_type(4)))  float    v4f;

#define WMMA_F16(A, B, C) \
    __builtin_amdgcn_wmma_f32_16x16x32_f16(false, (A), false, (B), (short)0, (C), false, false)

// Problem constants: B=16, N=4096, D=640, M=64
#define ROWS   65536   // B*N
#define DDIM   640
#define MDIM   64
#define KSTEPS 20      // DDIM / 32
#define DTILES 40      // DDIM / 16

__device__ __forceinline__ v16h cat8(v8h lo, v8h hi) {
    return __builtin_shufflevector(lo, hi, 0, 1, 2, 3, 4, 5, 6, 7,
                                           8, 9, 10, 11, 12, 13, 14, 15);
}

// Load 16 f32 in the CDNA5 16-bit A-matrix per-lane pattern (k..k+7, k+16..k+23)
// from global and convert to packed f16.
__device__ __forceinline__ v16h load_cvt_row(const float* __restrict__ p) {
    v4f q0 = *(const v4f*)(p + 0);
    v4f q1 = *(const v4f*)(p + 4);
    v4f q2 = *(const v4f*)(p + 16);
    v4f q3 = *(const v4f*)(p + 20);
    v16h r;
#pragma unroll
    for (int i = 0; i < 4; ++i) {
        r[i]      = (_Float16)q0[i];
        r[4 + i]  = (_Float16)q1[i];
        r[8 + i]  = (_Float16)q2[i];
        r[12 + i] = (_Float16)q3[i];
    }
    return r;
}

__global__ __launch_bounds__(256)
void ext_attn_kernel(const float* __restrict__ x,
                     const float* __restrict__ Mk,
                     const float* __restrict__ Mv,
                     float* __restrict__ out) {
    // 160 KB LDS total. s_mk is reused as the per-wave weight staging region
    // after GEMM1 (guarded by a block barrier).
    __shared__ _Float16 s_mk [MDIM * DDIM];  // 81920 B, XOR-16 chunk swizzle
    __shared__ _Float16 s_mvt[DDIM * MDIM];  // 81920 B, XOR-8  chunk swizzle

    const int tid    = threadIdx.x;
    const int wave   = tid >> 5;
    const int lane   = tid & 31;
    const int lane15 = lane & 15;
    const int kb     = (lane >> 4) << 3;     // K-half base: 0 (lanes 0-15) / 8 (16-31)
    const int cb     = kb >> 3;              // chunk base 0/1
    const int rbase  = (lane < 16) ? 0 : 8;  // C/D layout row base

    // ---- Stage Mk as f16 by 8-half chunks: 2x global b128 -> 1x ds b128 ----
    // Swizzle: chunk c=d>>3 stored at (c&~15)|((c^m)&15).
#pragma unroll 1
    for (int idx = tid; idx < MDIM * (DDIM / 8); idx += 256) {
        const int m = idx / (DDIM / 8);         // 80 chunks per row
        const int c = idx - m * (DDIM / 8);
        const int g = (c & ~15) | ((c ^ m) & 15);
        const float* src = Mk + m * DDIM + c * 8;
        v4f q0 = *(const v4f*)(src + 0);
        v4f q1 = *(const v4f*)(src + 4);
        v8h h;
#pragma unroll
        for (int i = 0; i < 4; ++i) {
            h[i]     = (_Float16)q0[i];
            h[4 + i] = (_Float16)q1[i];
        }
        *(v8h*)&s_mk[m * DDIM + g * 8] = h;
    }
    // ---- Stage Mv^T as f16 by float4: coalesced b128 reads, scalar transposed
    // stores. Swizzle: chunk c=m>>3 stored at c ^ (d&7) ^ ((d>>3)&1). ----
#pragma unroll 1
    for (int idx = tid; idx < MDIM * (DDIM / 4); idx += 256) {
        const int m  = idx / (DDIM / 4);        // 160 float4 per row
        const int d0 = (idx - m * (DDIM / 4)) * 4;
        v4f q = *(const v4f*)(Mv + m * DDIM + d0);
        const int cm = m >> 3, m7 = m & 7;
#pragma unroll
        for (int i = 0; i < 4; ++i) {
            const int d = d0 + i;
            const int g = cm ^ (d & 7) ^ ((d >> 3) & 1);
            s_mvt[d * MDIM + g * 8 + m7] = (_Float16)q[i];
        }
    }
    __syncthreads();

    const int tile = blockIdx.x * 8 + wave;   // gridDim.x == 4096/8
    const int row0 = tile * 16;
    const float* xrow = x + (size_t)(row0 + lane15) * DDIM;

    // ---------------- GEMM 1: scores[16 x 64] = x_tile @ Mk^T ----------------
    v8f c0 = {}, c1 = {}, c2 = {}, c3 = {};
#pragma unroll 4
    for (int ks = 0; ks < KSTEPS; ++ks) {
        v16h a = load_cvt_row(xrow + ks * 32 + kb);
        const int ca = ks * 4 + cb;      // first chunk index
        const int cc = ca + 2;           // +16 halves (same group of 16)
        const int hi4a = ca & ~15, lo4a = ca & 15;
        const int hi4c = cc & ~15, lo4c = cc & 15;
#pragma unroll
        for (int mt = 0; mt < 4; ++mt) {
            const int m = mt * 16 + lane15;
            const _Float16* mr = &s_mk[m * DDIM];
            v8h lo = *(const v8h*)&mr[(hi4a | (lo4a ^ lane15)) * 8];
            v8h hi = *(const v8h*)&mr[(hi4c | (lo4c ^ lane15)) * 8];
            v16h b = cat8(lo, hi);
            switch (mt) {
                case 0: c0 = WMMA_F16(a, b, c0); break;
                case 1: c1 = WMMA_F16(a, b, c1); break;
                case 2: c2 = WMMA_F16(a, b, c2); break;
                case 3: c3 = WMMA_F16(a, b, c3); break;
            }
        }
    }
    __syncthreads();   // everyone done reading s_mk before it becomes weight storage

    // ---------------- Softmax over M=64 per row ----------------
    // Weights written into (dead) s_mk region: per-wave 1024 halves, rows of 64
    // halves = 8 chunks, chunk swizzle c ^ (n&7) ^ ((n>>3)&1).
    _Float16* wbase = s_mk + wave * (16 * MDIM);
    const int m7 = lane15 & 7;
    const int cw = lane15 >> 3;   // column chunk base 0/1
#pragma unroll
    for (int r = 0; r < 8; ++r) {
        float s0 = c0[r], s1 = c1[r], s2 = c2[r], s3 = c3[r];
        float mx = fmaxf(fmaxf(s0, s1), fmaxf(s2, s3));
#pragma unroll
        for (int msk = 1; msk < 16; msk <<= 1)
            mx = fmaxf(mx, __shfl_xor(mx, msk, 32));
        float e0 = __expf(s0 - mx);
        float e1 = __expf(s1 - mx);
        float e2 = __expf(s2 - mx);
        float e3 = __expf(s3 - mx);
        float sum = e0 + e1 + e2 + e3;
#pragma unroll
        for (int msk = 1; msk < 16; msk <<= 1)
            sum += __shfl_xor(sum, msk, 32);
        float inv = __frcp_rn(sum);

        const int n   = rbase + r;
        const int nsw = (n & 7) ^ ((n >> 3) & 1);
        _Float16* wp = wbase + n * MDIM;
        wp[(((cw + 0) ^ nsw) << 3) + m7] = (_Float16)(e0 * inv);
        wp[(((cw + 2) ^ nsw) << 3) + m7] = (_Float16)(e1 * inv);
        wp[(((cw + 4) ^ nsw) << 3) + m7] = (_Float16)(e2 * inv);
        wp[(((cw + 6) ^ nsw) << 3) + m7] = (_Float16)(e3 * inv);
    }
    __syncthreads();   // order weight stores vs. A2 loads (uniform control flow)

    // ---------------- GEMM 2: out[16 x 640] = weights @ Mv ----------------
    const int nsw2 = (lane15 & 7) ^ ((lane15 >> 3) & 1);  // swizzle for row lane15
    const _Float16* wrow = wbase + lane15 * MDIM;
    v16h a2_0 = cat8(*(const v8h*)&wrow[(((cb + 0) ^ nsw2) << 3)],
                     *(const v8h*)&wrow[(((cb + 2) ^ nsw2) << 3)]);
    v16h a2_1 = cat8(*(const v8h*)&wrow[(((cb + 4) ^ nsw2) << 3)],
                     *(const v8h*)&wrow[(((cb + 6) ^ nsw2) << 3)]);

    float* obase = out + (size_t)(row0 + rbase) * DDIM + lane15;
#pragma unroll 2
    for (int dt = 0; dt < DTILES; ++dt) {
        // row d = dt*16 + lane15 -> same swizzle value nsw2 (dt*16 drops out)
        const _Float16* mvr = &s_mvt[(dt * 16 + lane15) * MDIM];
        v16h b2_0 = cat8(*(const v8h*)&mvr[(((cb + 0) ^ nsw2) << 3)],
                         *(const v8h*)&mvr[(((cb + 2) ^ nsw2) << 3)]);
        v16h b2_1 = cat8(*(const v8h*)&mvr[(((cb + 4) ^ nsw2) << 3)],
                         *(const v8h*)&mvr[(((cb + 6) ^ nsw2) << 3)]);
        v8f acc = {};
        acc = WMMA_F16(a2_0, b2_0, acc);
        acc = WMMA_F16(a2_1, b2_1, acc);
        float* op = obase + dt * 16;
#pragma unroll
        for (int r = 0; r < 8; ++r)
            op[(size_t)r * DDIM] = acc[r];
    }
}

extern "C" void kernel_launch(void* const* d_in, const int* in_sizes, int n_in,
                              void* d_out, int out_size, void* d_ws, size_t ws_size,
                              hipStream_t stream) {
    (void)in_sizes; (void)n_in; (void)out_size; (void)d_ws; (void)ws_size;
    const float* x  = (const float*)d_in[0];
    const float* Mk = (const float*)d_in[1];
    const float* Mv = (const float*)d_in[2];
    float* out = (float*)d_out;

    const int tiles  = ROWS / 16;   // 4096
    const int blocks = tiles / 8;   // 512 blocks x 8 waves = 1 tile/wave
    ext_attn_kernel<<<blocks, 256, 0, stream>>>(x, Mk, Mv, out);
}
